// RIPointTransformerAxesRegressor_4234837754423
// MI455X (gfx1250) — matrix operand order repaired and checked by reading, not compiled, not run on hardware
//
#include <hip/hip_runtime.h>
#include <hip/hip_bf16.h>
#include <math.h>

// ---------------------------------------------------------------------------
// CDNA5 (gfx1250) RIPointTransformer forward.
// Dense math: bf16 WMMA (v_wmma_f32_16x16x32_bf16), one wave per 16x16 C-tile,
// weights pre-packed into WMMA lane order (zero-padded) so the inner loop is
// branch-free: 4x b128 A loads + 2x b128 packed-B loads + 1 WMMA per K-step.
// ---------------------------------------------------------------------------

typedef __attribute__((ext_vector_type(16))) __bf16 v16bf;
typedef __attribute__((ext_vector_type(8)))  float  v8f;
typedef __attribute__((ext_vector_type(4)))  float  f32x4;

// ======================= W pack: fp32 [K,N] -> bf16 tiles ==================
// Layout: [nt][kt][lane][16 bf16]; lane l covers column n = nt*16+(l&15),
// k = kt*32 + (l>>4)*8 + (e<8 ? e : e+8)  for e=0..15.  Zero-padded past K/N edges.
__global__ void pack_w_kernel(const float* __restrict__ W, int K, int N,
                              __bf16* __restrict__ out)
{
    int ktiles = (K + 31) >> 5;
    int ntiles = (N + 15) >> 4;
    int t = blockIdx.x * blockDim.x + threadIdx.x;
    int total = ntiles * ktiles * 32;
    if (t >= total) return;
    int lane = t & 31;
    int kt   = (t >> 5) % ktiles;
    int nt   = (t >> 5) / ktiles;
    int n     = nt * 16 + (lane & 15);
    int kbase = kt * 32 + (lane >> 4) * 8;
    __bf16* o = out + (size_t)t * 16;
#pragma unroll
    for (int e = 0; e < 16; e++) {
        int k = kbase + (e < 8 ? e : e + 8);
        float v = (k < K && n < N) ? W[(size_t)k * N + n] : 0.0f;
        o[e] = (__bf16)v;
    }
}

// ============================ GEMM (bf16 WMMA) =============================
// C[M,N] = act( (acc? C : 0) + A_gathered[M,K] @ W[K,N] + bias[N] )
// Wp = packed weights (see pack_w_kernel). act: 0=none,1=relu,2=tanh.
__global__ void gemm_wmma_kernel(const float* __restrict__ A,
                                 const int*  __restrict__ gather,
                                 int lda,
                                 const __bf16* __restrict__ Wp,
                                 const float* __restrict__ bias,
                                 float* __restrict__ C,
                                 int M, int K, int N, int act, int acc)
{
    int wid  = (blockIdx.x * blockDim.x + threadIdx.x) >> 5;  // uniform per wave
    int lane = threadIdx.x & 31;
    int tilesN = (N + 15) >> 4;
    int tilesM = (M + 15) >> 4;
    if (wid >= tilesM * tilesN) return;
    int tm = wid / tilesN;
    int tn = wid % tilesN;
    int ktiles = (K + 31) >> 5;

    // A operand (16-bit A 16x32 layout): row = tile row + (lane&15),
    // lanes 0-15 take K 0..7 & 16..23, lanes 16-31 take K 8..15 & 24..31.
    int rowA = tm * 16 + (lane & 15);
    int gr   = rowA < M ? rowA : (M - 1);
    if (gather) gr = gather[gr];
    const float* arow = A + (size_t)gr * lda;
    int khalf = (lane >> 4) * 8;

    // Packed B: 16 contiguous bf16 per lane per ktile.
    const v16bf* bt = (const v16bf*)Wp + ((size_t)tn * ktiles) * 32 + lane;

    bool fullMN = ((tm * 16 + 16) <= M) && ((tn * 16 + 16) <= N);
    int  rbase  = tm * 16 + (lane >> 4) * 8;
    int  col    = tn * 16 + (lane & 15);

    v8f c;
    if (acc) {
        if (fullMN) {
#pragma unroll
            for (int i = 0; i < 8; i++) c[i] = C[(size_t)(rbase + i) * N + col];
        } else {
#pragma unroll
            for (int i = 0; i < 8; i++) {
                int r = rbase + i;
                c[i] = (r < M && col < N) ? C[(size_t)r * N + col] : 0.0f;
            }
        }
    } else {
#pragma unroll
        for (int i = 0; i < 8; i++) c[i] = 0.0f;
    }

    int kfull = K >> 5;  // lda==K in all call sites; K>=32 => lda%32==0 => 16B-aligned rows
    for (int kt = 0; kt < kfull; kt++) {
        const float* ap = arow + kt * 32 + khalf;
        __builtin_prefetch(ap + 32, 0, 1);                 // global_prefetch_b8
        f32x4 x0 = *(const f32x4*)(ap);
        f32x4 x1 = *(const f32x4*)(ap + 4);
        f32x4 x2 = *(const f32x4*)(ap + 16);
        f32x4 x3 = *(const f32x4*)(ap + 20);
        v16bf a;
#pragma unroll
        for (int e = 0; e < 4; e++) {
            a[e]      = (__bf16)x0[e];
            a[4 + e]  = (__bf16)x1[e];
            a[8 + e]  = (__bf16)x2[e];
            a[12 + e] = (__bf16)x3[e];
        }
        v16bf b = bt[(size_t)kt * 32];
        c = __builtin_amdgcn_wmma_f32_16x16x32_bf16(false, a, false, b,
                                                    (short)0, c, false, false);
    }

    if (K & 31) {  // remainder tile: clamp+select (v_cndmask), no EXEC branches
        int k0 = kfull * 32;
        v16bf a;
#pragma unroll
        for (int j = 0; j < 8; j++) {
            int ka = k0 + khalf + ((j < 4) ? (2 * j) : (16 + 2 * (j - 4)));
            float a0 = arow[(ka     < K) ? ka     : 0];
            float a1 = arow[(ka + 1 < K) ? ka + 1 : 0];
            a[2 * j]     = (__bf16)((ka     < K) ? a0 : 0.0f);
            a[2 * j + 1] = (__bf16)((ka + 1 < K) ? a1 : 0.0f);
        }
        v16bf b = bt[(size_t)kfull * 32];   // pack already zero-padded past K
        c = __builtin_amdgcn_wmma_f32_16x16x32_bf16(false, a, false, b,
                                                    (short)0, c, false, false);
    }

    if (fullMN) {
        float bb = bias[col];
#pragma unroll
        for (int i = 0; i < 8; i++) {
            float v = c[i] + bb;
            if (act == 1)      v = v > 0.0f ? v : 0.0f;
            else if (act == 2) v = tanhf(v);
            C[(size_t)(rbase + i) * N + col] = v;
        }
    } else {
#pragma unroll
        for (int i = 0; i < 8; i++) {
            int r = rbase + i;
            if (r < M && col < N) {
                float v = c[i] + bias[col];
                if (act == 1)      v = v > 0.0f ? v : 0.0f;
                else if (act == 2) v = tanhf(v);
                C[(size_t)r * N + col] = v;
            }
        }
    }
}

// ============================ FPS (single block) ===========================
__global__ void fps_kernel(const float* __restrict__ p, int N, int m,
                           int* __restrict__ idxs, float* __restrict__ dists)
{
    __shared__ float sd[1024];
    __shared__ int   si[1024];
    __shared__ int   slast;
    int tid = threadIdx.x;
    for (int j = tid; j < N; j += 1024) dists[j] = 1e10f;
    if (tid == 0) { idxs[0] = 0; slast = 0; }
    __syncthreads();

    for (int i = 1; i < m; i++) {
        int last = slast;
        float lx = p[last * 3], ly = p[last * 3 + 1], lz = p[last * 3 + 2];
        float best = -1.0f; int bi = 0x7fffffff;
        for (int j = tid; j < N; j += 1024) {
            float dx = p[j * 3] - lx, dy = p[j * 3 + 1] - ly, dz = p[j * 3 + 2] - lz;
            float dn = dx * dx + dy * dy + dz * dz;
            float dc = dists[j];
            dc = dn < dc ? dn : dc;
            dists[j] = dc;
            if (dc > best) { best = dc; bi = j; }
        }
        sd[tid] = best; si[tid] = bi;
        __syncthreads();
        for (int sft = 512; sft > 0; sft >>= 1) {
            if (tid < sft) {
                if (sd[tid + sft] > sd[tid] ||
                    (sd[tid + sft] == sd[tid] && si[tid + sft] < si[tid])) {
                    sd[tid] = sd[tid + sft]; si[tid] = si[tid + sft];
                }
            }
            __syncthreads();
        }
        if (tid == 0) { idxs[i] = si[0]; slast = si[0]; }
        __syncthreads();
    }
}

// ============================ kNN (brute force) ============================
template <int KN>
__global__ void knn_kernel(const float* __restrict__ q, int M,
                           const float* __restrict__ ref, int N,
                           int* __restrict__ out)
{
    int m = blockIdx.x * blockDim.x + threadIdx.x;
    if (m >= M) return;
    float qx = q[m * 3], qy = q[m * 3 + 1], qz = q[m * 3 + 2];
    float bd[KN]; int bi[KN];
#pragma unroll
    for (int i = 0; i < KN; i++) { bd[i] = 1e30f; bi[i] = 0; }
    for (int j = 0; j < N; j++) {
        float dx = ref[j * 3] - qx, dy = ref[j * 3 + 1] - qy, dz = ref[j * 3 + 2] - qz;
        float d = dx * dx + dy * dy + dz * dz;
        if (d < bd[KN - 1]) {
            int pos = KN - 1;
            while (pos > 0 && bd[pos - 1] > d) {
                bd[pos] = bd[pos - 1]; bi[pos] = bi[pos - 1]; pos--;
            }
            bd[pos] = d; bi[pos] = j;
        }
    }
    for (int i = 0; i < KN; i++) out[(size_t)m * KN + i] = bi[i];
}

// ============================ PPF features =================================
__device__ __forceinline__ float ppf_angle(float ax, float ay, float az,
                                           float bx, float by, float bz)
{
    float cx = ay * bz - az * by;
    float cy = az * bx - ax * bz;
    float cz = ax * by - ay * bx;
    float cn = sqrtf(cx * cx + cy * cy + cz * cz + 1e-12f);
    float dot = ax * bx + ay * by + az * bz;
    return atan2f(cn, dot);
}

__global__ void ppf_kernel(const float* __restrict__ pq, const float* __restrict__ nq,
                           const float* __restrict__ pref, const float* __restrict__ nref,
                           const int* __restrict__ gidx, float* __restrict__ ppf,
                           int M, int K)
{
    int t = blockIdx.x * blockDim.x + threadIdx.x;
    if (t >= M * K) return;
    int m = t / K;
    int r = gidx[t];
    float dx = pref[r * 3]     - pq[m * 3];
    float dy = pref[r * 3 + 1] - pq[m * 3 + 1];
    float dz = pref[r * 3 + 2] - pq[m * 3 + 2];
    float dist = sqrtf(dx * dx + dy * dy + dz * dz + 1e-12f);
    float n1x = nq[m * 3], n1y = nq[m * 3 + 1], n1z = nq[m * 3 + 2];
    float n2x = nref[r * 3], n2y = nref[r * 3 + 1], n2z = nref[r * 3 + 2];
    float* o = ppf + (size_t)t * 4;
    o[0] = ppf_angle(n1x, n1y, n1z, dx, dy, dz);
    o[1] = ppf_angle(n2x, n2y, n2z, dx, dy, dz);
    o[2] = ppf_angle(n1x, n1y, n1z, n2x, n2y, n2z);
    o[3] = dist;
}

// ============================ Local attention ==============================
template <int K>
__global__ void attn_kernel(const float* __restrict__ q,
                            const float* __restrict__ kb,
                            const float* __restrict__ vb,
                            float* __restrict__ y, int M, int h)
{
    int t = blockIdx.x * blockDim.x + threadIdx.x;
    if (t >= M * 4) return;
    int m = t >> 2, head = t & 3;
    int dh = h >> 2;
    float scale = rsqrtf((float)dh);
    const float* qp = q  + (size_t)m * h + head * dh;
    const float* kp = kb + (size_t)m * K * h + head * dh;
    const float* vp = vb + (size_t)m * K * h + head * dh;

    float logits[K];
    float mx = -1e30f;
#pragma unroll
    for (int k = 0; k < K; k++) {
        float s = 0.0f;
        for (int d = 0; d < dh; d++) s += qp[d] * kp[(size_t)k * h + d];
        s *= scale;
        logits[k] = s;
        mx = s > mx ? s : mx;
    }
    float sum = 0.0f;
#pragma unroll
    for (int k = 0; k < K; k++) { float e = expf(logits[k] - mx); logits[k] = e; sum += e; }
    float inv = 1.0f / sum;
    float* yp = y + (size_t)m * h + head * dh;
    for (int d = 0; d < dh; d++) {
        float o = 0.0f;
#pragma unroll
        for (int k = 0; k < K; k++) o += logits[k] * vp[(size_t)k * h + d];
        yp[d] = o * inv;
    }
}

// ==================== LayerNorm + residual + ReLU ==========================
__global__ void ln_res_relu_kernel(const float* __restrict__ yin,
                                   const float* __restrict__ g,
                                   const float* __restrict__ b,
                                   float* __restrict__ x, int M, int C)
{
    int m = blockIdx.x * blockDim.x + threadIdx.x;
    if (m >= M) return;
    const float* yp = yin + (size_t)m * C;
    float* xp = x + (size_t)m * C;
    float mu = 0.0f;
    for (int c = 0; c < C; c++) mu += yp[c];
    mu /= (float)C;
    float var = 0.0f;
    for (int c = 0; c < C; c++) { float d = yp[c] - mu; var += d * d; }
    var /= (float)C;
    float inv = rsqrtf(var + 1e-5f);
    for (int c = 0; c < C; c++) {
        float v = (yp[c] - mu) * inv * g[c] + b[c] + xp[c];
        xp[c] = v > 0.0f ? v : 0.0f;
    }
}

// ============================ gather [M,3] =================================
__global__ void gather3_kernel(const float* __restrict__ src,
                               const int* __restrict__ idx,
                               float* __restrict__ dst, int M)
{
    int t = blockIdx.x * blockDim.x + threadIdx.x;
    if (t >= M * 3) return;
    int m = t / 3, c = t - m * 3;
    dst[t] = src[(size_t)idx[m] * 3 + c];
}

// ===========================================================================
// Host-side orchestration
// ===========================================================================

static const int PLANES_[5]    = {32, 64, 128, 256, 512};
static const int STRIDES_[5]   = {1, 4, 4, 4, 4};
static const int NSAMPLE_[5]   = {36, 24, 24, 24, 24};
static const int NBLOCKS_[5]   = {2, 3, 4, 6, 3};
static const int REG_DIMS_[10] = {512, 512, 1024, 1024, 512, 256, 128, 64, 32, 3};

struct BlockP { int Wq, bq, Wk, bk, Wv, bv, Wp1, bp1, Wp2, bp2, Wo, bo, g, b; };

static void launch_gemm(const float* A, const int* gather, int lda,
                        const float* W, const float* bias, float* C,
                        int M, int K, int N, int act, int acc,
                        __bf16* wpack, hipStream_t s)
{
    int ktiles = (K + 31) / 32;
    int ntiles = (N + 15) / 16;
    int packT  = ktiles * ntiles * 32;
    pack_w_kernel<<<(packT + 255) / 256, 256, 0, s>>>(W, K, N, wpack);
    int tiles  = ((M + 15) / 16) * ntiles;
    gemm_wmma_kernel<<<(tiles + 3) / 4, 128, 0, s>>>(A, gather, lda, wpack, bias,
                                                     C, M, K, N, act, acc);
}

static void launch_knn(const float* q, int M, const float* ref, int N,
                       int* out, int k, hipStream_t s)
{
    int blocks = (M + 255) / 256;
    if (k == 36) knn_kernel<36><<<blocks, 256, 0, s>>>(q, M, ref, N, out);
    else         knn_kernel<24><<<blocks, 256, 0, s>>>(q, M, ref, N, out);
}

static void launch_attn(const float* q, const float* kb, const float* vb,
                        float* y, int M, int k, int h, hipStream_t s)
{
    int blocks = (M * 4 + 255) / 256;
    if (k == 36) attn_kernel<36><<<blocks, 256, 0, s>>>(q, kb, vb, y, M, h);
    else         attn_kernel<24><<<blocks, 256, 0, s>>>(q, kb, vb, y, M, h);
}

extern "C" void kernel_launch(void* const* d_in, const int* in_sizes, int n_in,
                              void* d_out, int out_size, void* d_ws, size_t ws_size,
                              hipStream_t stream)
{
    (void)in_sizes; (void)n_in; (void)out_size; (void)ws_size;

    // ---- inputs (setup_inputs dict order) ----
    const float* p0 = (const float*)d_in[0];   // [8192,3]
    const float* x0 = (const float*)d_in[1];   // [8192,1]
    // d_in[2] = o0 (unused)
    const float* n0 = (const float*)d_in[3];   // [8192,3]

    // ---- parameter index table (make_params insertion order, starting at 4) ----
    BlockP bp[5][6];
    int cnt = 4;
    for (int s = 0; s < 5; s++) {
        for (int j = 0; j < NBLOCKS_[s]; j++) {
            BlockP& P = bp[s][j];
            P.Wq = cnt++; P.bq = cnt++;
            P.Wk = cnt++; P.bk = cnt++;
            P.Wv = cnt++; P.bv = cnt++;
            P.Wp1 = cnt++; P.bp1 = cnt++;
            P.Wp2 = cnt++; P.bp2 = cnt++;
            P.Wo = cnt++; P.bo = cnt++;
            if (j > 0) { P.g = cnt++; P.b = cnt++; } else { P.g = -1; P.b = -1; }
        }
    }
    int regW[9], regb[9];
    for (int i = 0; i < 9; i++) { regW[i] = cnt++; regb[i] = cnt++; }
    auto F = [&](int i) -> const float* { return (const float*)d_in[i]; };

    // ---- workspace bump allocator ----
    char* base = (char*)d_ws;
    size_t off = 0;
    auto alloc = [&](size_t bytes) -> char* {
        char* p = base + off;
        off = (off + bytes + 255) & ~(size_t)255;
        return p;
    };
    float*  pA    = (float*)alloc((size_t)8192 * 3 * 4);
    float*  pB    = (float*)alloc((size_t)8192 * 3 * 4);
    float*  nA    = (float*)alloc((size_t)8192 * 3 * 4);
    float*  nB    = (float*)alloc((size_t)8192 * 3 * 4);
    int*    idxb  = (int*)  alloc((size_t)8192 * 4);
    float*  fpsd  = (float*)alloc((size_t)8192 * 4);
    int*    gidx  = (int*)  alloc((size_t)8192 * 36 * 4);
    int*    gidx2 = (int*)  alloc((size_t)2048 * 36 * 4);
    float*  ppfb  = (float*)alloc((size_t)8192 * 36 * 4 * 4);
    float*  ppf2  = (float*)alloc((size_t)2048 * 36 * 4 * 4);
    size_t  big   = (size_t)8192 * 36 * 32;            // max M*K*h rows (stage 0)
    float*  keyb  = (float*)alloc(big * 4);
    float*  vbuf  = (float*)alloc(big * 4);
    float*  pehb  = (float*)alloc(big * 4);
    float*  qb    = (float*)alloc((size_t)8192 * 32 * 4);
    float*  yb    = (float*)alloc((size_t)8192 * 32 * 4);
    float*  xT    = (float*)alloc((size_t)8192 * 32 * 4);
    float*  xA    = (float*)alloc((size_t)8192 * 32 * 4);
    float*  xB    = (float*)alloc((size_t)8192 * 32 * 4);
    __bf16* wpack = (__bf16*)alloc((size_t)32 * 64 * 32 * 16 * 2 + 4096); // max ktiles*ntiles*32*16 bf16

    // ---- stage loop ----
    const float* pcur = p0;
    const float* ncur = n0;
    const float* xcur = x0;
    int Ccur = 1, Ncur = 8192;

    for (int s = 0; s < 5; s++) {
        int out = PLANES_[s];
        int h   = out > 256 ? 256 : out;
        int kk  = NSAMPLE_[s];
        int nb  = NBLOCKS_[s];
        int stride = STRIDES_[s];
        int M = Ncur / stride;

        const float *np_, *nn_;
        const int* idxp = nullptr;
        if (stride != 1) {
            fps_kernel<<<1, 1024, 0, stream>>>(pcur, Ncur, M, idxb, fpsd);
            float* pd = (pcur == pA) ? pB : pA;
            float* nd = (ncur == nA) ? nB : nA;
            int bl = (M * 3 + 255) / 256;
            gather3_kernel<<<bl, 256, 0, stream>>>(pcur, idxb, pd, M);
            gather3_kernel<<<bl, 256, 0, stream>>>(ncur, idxb, nd, M);
            np_ = pd; nn_ = nd; idxp = idxb;
        } else {
            np_ = pcur; nn_ = ncur;
        }

        launch_knn(np_, M, pcur, Ncur, gidx, kk, stream);
        {
            int bl = (M * kk + 255) / 256;
            ppf_kernel<<<bl, 256, 0, stream>>>(np_, nn_, pcur, ncur, gidx, ppfb, M, kk);
        }

        // ---- first block: cross-attention (q from centers, k/v from neighbors) ----
        const BlockP& P0 = bp[s][0];
        launch_gemm(xcur, gidx, Ccur, F(P0.Wk), F(P0.bk), keyb, M * kk, Ccur, h, 0, 0, wpack, stream);
        launch_gemm(xcur, gidx, Ccur, F(P0.Wv), F(P0.bv), vbuf, M * kk, Ccur, h, 0, 0, wpack, stream);
        launch_gemm(ppfb, nullptr, 4, F(P0.Wp1), F(P0.bp1), pehb, M * kk, 4, h, 1, 0, wpack, stream);
        launch_gemm(pehb, nullptr, h, F(P0.Wp2), F(P0.bp2), keyb, M * kk, h, h, 0, 1, wpack, stream);
        launch_gemm(pehb, nullptr, h, F(P0.Wp2), F(P0.bp2), vbuf, M * kk, h, h, 0, 1, wpack, stream);
        launch_gemm(xcur, idxp, Ccur, F(P0.Wq), F(P0.bq), qb, M, Ccur, h, 0, 0, wpack, stream);
        launch_attn(qb, keyb, vbuf, yb, M, kk, h, stream);
        float* xnext = (xcur == xA) ? xB : xA;
        launch_gemm(yb, nullptr, h, F(P0.Wo), F(P0.bo), xnext, M, h, out, 0, 0, wpack, stream);
        float* xw = xnext;
        Ccur = out;

        // ---- neighbor structure for residual blocks ----
        const int* g2; const float* pp2;
        if (stride != 1) {
            launch_knn(np_, M, np_, M, gidx2, kk, stream);
            int bl = (M * kk + 255) / 256;
            ppf_kernel<<<bl, 256, 0, stream>>>(np_, nn_, np_, nn_, gidx2, ppf2, M, kk);
            g2 = gidx2; pp2 = ppf2;
        } else {
            g2 = gidx; pp2 = ppfb;
        }

        // ---- residual blocks ----
        for (int j = 1; j < nb; j++) {
            const BlockP& Pj = bp[s][j];
            launch_gemm(xw, g2, Ccur, F(Pj.Wk), F(Pj.bk), keyb, M * kk, Ccur, h, 0, 0, wpack, stream);
            launch_gemm(xw, g2, Ccur, F(Pj.Wv), F(Pj.bv), vbuf, M * kk, Ccur, h, 0, 0, wpack, stream);
            launch_gemm(pp2, nullptr, 4, F(Pj.Wp1), F(Pj.bp1), pehb, M * kk, 4, h, 1, 0, wpack, stream);
            launch_gemm(pehb, nullptr, h, F(Pj.Wp2), F(Pj.bp2), keyb, M * kk, h, h, 0, 1, wpack, stream);
            launch_gemm(pehb, nullptr, h, F(Pj.Wp2), F(Pj.bp2), vbuf, M * kk, h, h, 0, 1, wpack, stream);
            launch_gemm(xw, nullptr, Ccur, F(Pj.Wq), F(Pj.bq), qb, M, Ccur, h, 0, 0, wpack, stream);
            launch_attn(qb, keyb, vbuf, yb, M, kk, h, stream);
            launch_gemm(yb, nullptr, h, F(Pj.Wo), F(Pj.bo), xT, M, h, out, 0, 0, wpack, stream);
            int bl = (M + 127) / 128;
            ln_res_relu_kernel<<<bl, 128, 0, stream>>>(xT, F(Pj.g), F(Pj.b), xw, M, out);
        }

        xcur = xw; pcur = np_; ncur = nn_; Ncur = M;
    }

    // ---- AxesRegressor: 9 linear layers on [32, .] ----
    const float* rin = xcur;
    for (int i = 0; i < 9; i++) {
        int fin = REG_DIMS_[i], fout = REG_DIMS_[i + 1];
        float* dst = (i == 8) ? (float*)d_out : ((i & 1) ? yb : qb);
        launch_gemm(rin, nullptr, fin, F(regW[i]), F(regb[i]), dst,
                    32, fin, fout, (i == 8) ? 2 : 1, 0, wpack, stream);
        rin = dst;
    }
}
